// Block_2680059593149
// MI455X (gfx1250) — matrix operand order repaired and checked by reading, not compiled, not run on hardware
//
#include <hip/hip_runtime.h>
#include <cstdint>

// ---------------------------------------------------------------------------
// Types for CDNA5 WMMA (wave32): v_wmma_f32_16x16x32_bf16
// ---------------------------------------------------------------------------
typedef __bf16 bf16;
typedef __attribute__((ext_vector_type(16))) __bf16 v16bf;
typedef __attribute__((ext_vector_type(8)))  __bf16 v8bf;
typedef __attribute__((ext_vector_type(8)))  float  v8f;

static_assert(sizeof(bf16) == 2, "bf16 must be 2 bytes");

#define E_DIM 1024
#define T_DIM 1024
#define B_DIM 8
#define H_DIM 16
#define D_DIM 64
#define NTOK  (B_DIM * T_DIM)   // 8192

__device__ __forceinline__ v8f v8f_zero() {
  v8f z;
#pragma unroll
  for (int i = 0; i < 8; ++i) z[i] = 0.0f;
  return z;
}

// A-fragment (16-bit A 16x32): lane = M row (lane&15).
// element e: K = half*16 + e (matches B fragment order)
// -> two contiguous 8-element (16B) loads per lane.
__device__ __forceinline__ v16bf load_a_frag(const bf16* rowPtr, int k0, int half) {
  const v8bf lo = *reinterpret_cast<const v8bf*>(rowPtr + k0 + half * 16);
  const v8bf hi = *reinterpret_cast<const v8bf*>(rowPtr + k0 + half * 16 + 8);
  v16bf f;
#pragma unroll
  for (int i = 0; i < 8; ++i) { f[i] = lo[i]; f[i + 8] = hi[i]; }
  return f;
}

// B-fragment (16-bit B 32x16): lane = N col (lane&15); element e: K = half*16 + e.
// colPtr = &W[n][0] (row-major [out,in]) -> one contiguous 16-element (32B) load.
__device__ __forceinline__ v16bf load_b_frag(const bf16* colPtr, int k0, int half) {
  return *reinterpret_cast<const v16bf*>(colPtr + k0 + half * 16);
}

// B-fragment with strided K (used for V: [token, d] layout, fixed d, varying token)
__device__ __forceinline__ v16bf load_b_frag_strided(const bf16* base, int stride) {
  v16bf f;
#pragma unroll
  for (int e = 0; e < 16; ++e) f[e] = base[(size_t)e * stride];
  return f;
}

// ---------------------------------------------------------------------------
// fp32 -> bf16 conversion (weights)
// ---------------------------------------------------------------------------
__global__ void f32_to_bf16(const float* __restrict__ src, bf16* __restrict__ dst, int n) {
  int i = blockIdx.x * blockDim.x + threadIdx.x;
  if (i < n) dst[i] = (bf16)src[i];
}

// ---------------------------------------------------------------------------
// LayerNorm over E=1024, one row per block (256 threads, 4 elems/thread),
// bf16 output feeding the WMMA GEMMs.
// ---------------------------------------------------------------------------
__global__ __launch_bounds__(256) void ln_bf16_kernel(
    const float* __restrict__ x, const float* __restrict__ g,
    const float* __restrict__ bta, bf16* __restrict__ out) {
  const int row = blockIdx.x;
  const float* xr = x + (size_t)row * E_DIM;
  float s = 0.f, s2 = 0.f;
#pragma unroll
  for (int i = 0; i < 4; ++i) {
    float v = xr[threadIdx.x + i * 256];
    s += v; s2 += v * v;
  }
  __shared__ float rs[256], rs2[256];
  rs[threadIdx.x] = s; rs2[threadIdx.x] = s2;
  __syncthreads();
  for (int off = 128; off > 0; off >>= 1) {
    if (threadIdx.x < off) {
      rs[threadIdx.x]  += rs[threadIdx.x + off];
      rs2[threadIdx.x] += rs2[threadIdx.x + off];
    }
    __syncthreads();
  }
  const float mean = rs[0] * (1.0f / E_DIM);
  const float var  = rs2[0] * (1.0f / E_DIM) - mean * mean;
  const float rstd = rsqrtf(var + 1e-5f);
  bf16* orow = out + (size_t)row * E_DIM;
#pragma unroll
  for (int i = 0; i < 4; ++i) {
    int c = threadIdx.x + i * 256;
    orow[c] = (bf16)((xr[c] - mean) * rstd * g[c] + bta[c]);
  }
}

// ---------------------------------------------------------------------------
// WMMA GEMM: out[M,N] = A[M,K] @ W[N,K]^T (+ epilogue)
// 8 waves/block: wave tile 32(M) x 64(N) -> 2 A-frags x 4 B-frags = 8 WMMA/K-step.
// Block tile 64(M) x 256(N). M=8192; N in {1024,3072,4096}; K in {1024,4096}.
// ---------------------------------------------------------------------------
enum { EPI_BF16 = 0, EPI_BIAS_RES_F32 = 1, EPI_BIAS_RELU_BF16 = 2 };

template <int EPI>
__global__ __launch_bounds__(256) void gemm_bf16_wmma(
    const bf16* __restrict__ A, const bf16* __restrict__ W,
    void* __restrict__ dst, const float* __restrict__ bias,
    const float* __restrict__ res, int M, int N, int K) {
  const int lane = threadIdx.x & 31;
  const int half = lane >> 4;
  const int wave = threadIdx.x >> 5;
  const int wm = wave >> 2;   // 0..1
  const int wn = wave & 3;    // 0..3
  const int rowTile = blockIdx.x * 64 + wm * 32;   // 32 M-rows per wave
  const int colTile = blockIdx.y * 256 + wn * 64;  // 64 N-cols per wave

  const bf16* arow0 = A + (size_t)(rowTile + (lane & 15)) * K;
  const bf16* arow1 = arow0 + (size_t)16 * K;
  const bf16* wrow[4];
#pragma unroll
  for (int j = 0; j < 4; ++j)
    wrow[j] = W + (size_t)(colTile + j * 16 + (lane & 15)) * K;

  v8f acc0[4], acc1[4];
#pragma unroll
  for (int j = 0; j < 4; ++j) { acc0[j] = v8f_zero(); acc1[j] = v8f_zero(); }

  for (int k0 = 0; k0 < K; k0 += 32) {
    if (k0 + 64 < K) {
      __builtin_prefetch(arow0 + k0 + 64, 0, 1);  // global_prefetch_b8
      __builtin_prefetch(arow1 + k0 + 64, 0, 1);
    }
    v16bf a0 = load_a_frag(arow0, k0, half);
    v16bf a1 = load_a_frag(arow1, k0, half);
#pragma unroll
    for (int j = 0; j < 4; ++j) {
      v16bf bw = load_b_frag(wrow[j], k0, half);
      acc0[j] = __builtin_amdgcn_wmma_f32_16x16x32_bf16(
          false, a0, false, bw, (short)0, acc0[j], false, false);
      acc1[j] = __builtin_amdgcn_wmma_f32_16x16x32_bf16(
          false, a1, false, bw, (short)0, acc1[j], false, false);
    }
  }

  // C/D layout (ISA 7.12.2): lane -> col = colTile + (lane&15); VGPR r -> row r + 8*half
#pragma unroll
  for (int j = 0; j < 4; ++j) {
    const int col = colTile + j * 16 + (lane & 15);
#pragma unroll
    for (int r = 0; r < 8; ++r) {
#pragma unroll
      for (int t = 0; t < 2; ++t) {
        const int row = rowTile + t * 16 + r + 8 * half;
        const size_t idx = (size_t)row * N + col;
        const float v = (t == 0) ? acc0[j][r] : acc1[j][r];
        if constexpr (EPI == EPI_BF16) {
          ((bf16*)dst)[idx] = (bf16)v;
        } else if constexpr (EPI == EPI_BIAS_RES_F32) {
          ((float*)dst)[idx] = v + bias[col] + res[idx];
        } else {
          ((bf16*)dst)[idx] = (bf16)fmaxf(v + bias[col], 0.0f);
        }
      }
    }
  }
}

// ---------------------------------------------------------------------------
// Causal flash attention, one wave per (b, h, 16-query tile).
// qkv: bf16 [NTOK, 3072] ; cols [0,1024)=Q, [1024,2048)=K, [2048,3072)=V,
// each packed per-head as h*64+d. Key blocks of 32 so P(16x32) is one A-frag.
// Row-sum l is tracked as a 5th WMMA accumulator (P x ones): its C layout
// replicates the row-sum across all 16 lanes of each half, so only the
// row-max needs a cross-lane shuffle reduction.
// scale = E^-0.5 = 1/32 (reference scales by n_embd, not head_size).
// ---------------------------------------------------------------------------
__global__ __launch_bounds__(32) void attn_kernel(
    const bf16* __restrict__ qkv, bf16* __restrict__ attn) {
  const int lane = threadIdx.x & 31;
  const int half = lane >> 4;
  const int n16  = lane & 15;
  const int qt = blockIdx.x;   // query tile 0..63
  const int h  = blockIdx.y;   // head
  const int b  = blockIdx.z;   // batch
  const int qbase = qt * 16;
  const int LDQ = 3 * E_DIM;   // 3072
  const bf16* base = qkv + (size_t)b * T_DIM * LDQ;
  const int qoff = h * D_DIM;
  const int koff = E_DIM + h * D_DIM;
  const int voff = 2 * E_DIM + h * D_DIM;
  const float scale = 0.03125f;  // 1024^-0.5

  // Q fragments (kept in registers for whole loop): K-dim = D = 64 -> 2 frags
  const bf16* qrow = base + (size_t)(qbase + n16) * LDQ + qoff;
  const v16bf qa0 = load_a_frag(qrow, 0, half);
  const v16bf qa1 = load_a_frag(qrow, 32, half);

  // all-ones B fragment for the row-sum accumulator tile
  v16bf ones;
#pragma unroll
  for (int e = 0; e < 16; ++e) ones[e] = (bf16)1.0f;

  v8f o[4];
#pragma unroll
  for (int t = 0; t < 4; ++t) o[t] = v8f_zero();
  v8f lacc = v8f_zero();
  float mstate[8];
#pragma unroll
  for (int r = 0; r < 8; ++r) mstate[r] = -__builtin_inff();

  __shared__ alignas(32) bf16 lds_p[16 * 32];

  for (int s0 = 0; s0 < qbase + 16; s0 += 32) {
    // ---- S = Q K^T for 32 keys: two 16-wide C tiles ----
    v8f st[2];
#pragma unroll
    for (int j = 0; j < 2; ++j) {
      st[j] = v8f_zero();
      const bf16* kcol = base + (size_t)(s0 + j * 16 + n16) * LDQ + koff;
      v16bf kb0 = load_b_frag(kcol, 0, half);
      v16bf kb1 = load_b_frag(kcol, 32, half);
      st[j] = __builtin_amdgcn_wmma_f32_16x16x32_bf16(
          false, qa0, false, kb0, (short)0, st[j], false, false);
      st[j] = __builtin_amdgcn_wmma_f32_16x16x32_bf16(
          false, qa1, false, kb1, (short)0, st[j], false, false);
    }

    // ---- scale + causal mask + per-row max ----
    float rowmax[8];
#pragma unroll
    for (int r = 0; r < 8; ++r) rowmax[r] = -__builtin_inff();
#pragma unroll
    for (int j = 0; j < 2; ++j) {
      const int key = s0 + j * 16 + n16;
#pragma unroll
      for (int r = 0; r < 8; ++r) {
        const int q = qbase + r + 8 * half;
        float v = st[j][r] * scale;
        v = (key <= q) ? v : -__builtin_inff();
        st[j][r] = v;
        rowmax[r] = fmaxf(rowmax[r], v);
      }
    }
    // row m = r + 8*half lives in the 16-lane group "half" -> xor-reduce over 1,2,4,8
#pragma unroll
    for (int off = 1; off < 16; off <<= 1)
#pragma unroll
      for (int r = 0; r < 8; ++r)
        rowmax[r] = fmaxf(rowmax[r], __shfl_xor(rowmax[r], off, 32));

    // ---- online softmax update ----
    float rescale[8];
#pragma unroll
    for (int r = 0; r < 8; ++r) {
      const float mnew = fmaxf(mstate[r], rowmax[r]);
      rescale[r] = __expf(mstate[r] - mnew);  // first iter: exp(-inf)=0
      mstate[r] = mnew;
    }
#pragma unroll
    for (int j = 0; j < 2; ++j) {
#pragma unroll
      for (int r = 0; r < 8; ++r) {
        const float p = __expf(st[j][r] - mstate[r]);  // masked -> exp(-inf)=0
        lds_p[(r + 8 * half) * 32 + j * 16 + n16] = (bf16)p;
      }
    }
    // rescale running accumulators (including row-sum tile) BEFORE adding block
#pragma unroll
    for (int t = 0; t < 4; ++t)
#pragma unroll
      for (int r = 0; r < 8; ++r)
        o[t][r] *= rescale[r];
#pragma unroll
    for (int r = 0; r < 8; ++r)
      lacc[r] *= rescale[r];

    __syncthreads();  // P written to LDS before A-frag reads
    const v16bf pa = load_a_frag(lds_p + n16 * 32, 0, half);

    // ---- O += P(16x32) x V(32x64): 4 WMMAs over D tiles; l += P x ones ----
#pragma unroll
    for (int t = 0; t < 4; ++t) {
      const bf16* vb = base + (size_t)(s0 + half * 16) * LDQ + voff + t * 16 + n16;
      v16bf vf = load_b_frag_strided(vb, LDQ);
      o[t] = __builtin_amdgcn_wmma_f32_16x16x32_bf16(
          false, pa, false, vf, (short)0, o[t], false, false);
    }
    lacc = __builtin_amdgcn_wmma_f32_16x16x32_bf16(
        false, pa, false, ones, (short)0, lacc, false, false);
    __syncthreads();  // LDS reads complete before next iteration overwrites
  }

  // ---- normalize and store heads-concatenated [NTOK, E] as bf16 ----
  bf16* abase = attn + (size_t)b * T_DIM * E_DIM;
#pragma unroll
  for (int t = 0; t < 4; ++t) {
    const int col = h * D_DIM + t * 16 + n16;
#pragma unroll
    for (int r = 0; r < 8; ++r) {
      const int row = qbase + r + 8 * half;
      abase[(size_t)row * E_DIM + col] = (bf16)(o[t][r] * (1.0f / lacc[r]));
    }
  }
}

// ---------------------------------------------------------------------------
// Launch sequence
// ---------------------------------------------------------------------------
extern "C" void kernel_launch(void* const* d_in, const int* in_sizes, int n_in,
                              void* d_out, int out_size, void* d_ws, size_t ws_size,
                              hipStream_t stream) {
  (void)in_sizes; (void)n_in; (void)out_size; (void)ws_size;
  const float* x      = (const float*)d_in[0];
  const float* wq     = (const float*)d_in[1];
  const float* wk     = (const float*)d_in[2];
  const float* wv     = (const float*)d_in[3];
  const float* w_proj = (const float*)d_in[4];
  const float* b_proj = (const float*)d_in[5];
  const float* w1     = (const float*)d_in[6];
  const float* b1     = (const float*)d_in[7];
  const float* w2     = (const float*)d_in[8];
  const float* b2     = (const float*)d_in[9];
  const float* ln1_g  = (const float*)d_in[10];
  const float* ln1_b  = (const float*)d_in[11];
  const float* ln2_g  = (const float*)d_in[12];
  const float* ln2_b  = (const float*)d_in[13];
  float* out = (float*)d_out;

  char* ws = (char*)d_ws;
  bf16*  h1   = (bf16*)(ws);                         // 16 MB  ln1(x) bf16 [8192,1024]
  bf16*  wqkv = (bf16*)(ws + ((size_t)16  << 20));   //  6 MB  [3072,1024]
  bf16*  qkv  = (bf16*)(ws + ((size_t)22  << 20));   // 48 MB  [8192,3072]
  bf16*  attn = (bf16*)(ws + ((size_t)70  << 20));   // 16 MB  [8192,1024]
  bf16*  wpj  = (bf16*)(ws + ((size_t)86  << 20));   //  2 MB  [1024,1024]
  float* x1   = (float*)(ws + ((size_t)88  << 20));  // 32 MB  x + sa, fp32
  bf16*  h2   = (bf16*)(ws + ((size_t)120 << 20));   // 16 MB  ln2(x1) bf16
  bf16*  w1b  = (bf16*)(ws + ((size_t)136 << 20));   //  8 MB  [4096,1024]
  bf16*  w2b  = (bf16*)(ws + ((size_t)144 << 20));   //  8 MB  [1024,4096]
  bf16*  a1   = (bf16*)(ws + ((size_t)152 << 20));   // 64 MB  relu(ffn1) [8192,4096]

  // Weight down-conversion fp32 -> bf16 (48 MB fp32, negligible)
  const int NW = E_DIM * E_DIM;  // 1M
  f32_to_bf16<<<NW / 256, 256, 0, stream>>>(wq, wqkv, NW);
  f32_to_bf16<<<NW / 256, 256, 0, stream>>>(wk, wqkv + NW, NW);
  f32_to_bf16<<<NW / 256, 256, 0, stream>>>(wv, wqkv + 2 * NW, NW);
  f32_to_bf16<<<NW / 256, 256, 0, stream>>>(w_proj, wpj, NW);
  f32_to_bf16<<<(4 * NW) / 256, 256, 0, stream>>>(w1, w1b, 4 * NW);
  f32_to_bf16<<<(4 * NW) / 256, 256, 0, stream>>>(w2, w2b, 4 * NW);

  // h1 = ln1(x)
  ln_bf16_kernel<<<NTOK, 256, 0, stream>>>(x, ln1_g, ln1_b, h1);

  // QKV: [8192,1024] @ [3072,1024]^T -> [8192,3072] bf16
  gemm_bf16_wmma<EPI_BF16><<<dim3(NTOK / 64, 3072 / 256), 256, 0, stream>>>(
      h1, wqkv, qkv, nullptr, nullptr, NTOK, 3072, E_DIM);

  // causal attention per (b,h,qtile)
  attn_kernel<<<dim3(T_DIM / 16, H_DIM, B_DIM), 32, 0, stream>>>(qkv, attn);

  // x1 = x + attn @ w_proj^T + b_proj   (fp32)
  gemm_bf16_wmma<EPI_BIAS_RES_F32><<<dim3(NTOK / 64, E_DIM / 256), 256, 0, stream>>>(
      attn, wpj, x1, b_proj, x, NTOK, E_DIM, E_DIM);

  // h2 = ln2(x1)
  ln_bf16_kernel<<<NTOK, 256, 0, stream>>>(x1, ln2_g, ln2_b, h2);

  // a1 = relu(h2 @ w1^T + b1)  -> bf16 [8192,4096]
  gemm_bf16_wmma<EPI_BIAS_RELU_BF16><<<dim3(NTOK / 64, 4096 / 256), 256, 0, stream>>>(
      h2, w1b, a1, b1, nullptr, NTOK, 4096, E_DIM);

  // out = x1 + a1 @ w2^T + b2  (fp32, final output)
  gemm_bf16_wmma<EPI_BIAS_RES_F32><<<dim3(NTOK / 64, E_DIM / 256), 256, 0, stream>>>(
      a1, w2b, out, b2, x1, NTOK, E_DIM, 4096);
}
